// GraphTemporalLayer_3633542332534
// MI455X (gfx1250) — matrix operand order repaired and checked by reading, not compiled, not run on hardware
//
#include <hip/hip_runtime.h>

#define NB 4096
#define G  2                 // batch elements per workgroup (M = 16*G)
#define NWG (NB / G)
#define DD 512
#define NH 8
#define DH 64

typedef _Float16 half16 __attribute__((ext_vector_type(16)));
typedef _Float16 half8  __attribute__((ext_vector_type(8)));
typedef float    v8f    __attribute__((ext_vector_type(8)));

// ---------------------------------------------------------------------------
// Kernel 0: convert weights to f16 in WMMA-B-friendly [n][k] layout.
//   [0] QwT  [1] KwT  [2] VwT  (transposed: out[n][k] = W[k][n])
//   [3] ff1  (already [n][k] since h = y @ ff1_w.T)
// ---------------------------------------------------------------------------
__global__ void convert_weights_kernel(const float* __restrict__ Qw,
                                       const float* __restrict__ Kw,
                                       const float* __restrict__ Vw,
                                       const float* __restrict__ ff1,
                                       _Float16* __restrict__ ws) {
  int idx = blockIdx.x * 256 + threadIdx.x;     // 4 * 512 * 512 total
  int m = idx >> 18;
  int r = idx & ((1 << 18) - 1);
  if (m < 3) {
    const float* src = (m == 0) ? Qw : (m == 1) ? Kw : Vw;
    int k = r >> 9, n = r & 511;                // coalesced reads over n
    ws[(size_t)m * 262144 + (size_t)n * 512 + k] = (_Float16)src[(size_t)k * 512 + n];
  } else {
    ws[(size_t)3 * 262144 + r] = (_Float16)ff1[r];
  }
}

// WMMA f16 A-operand: lane holds row m=lane%16; halves [0..7] at K=g*8,
// halves [8..15] at K=16+g*8 (g = lane/16). p points at the K=g*8 chunk.
__device__ __forceinline__ half16 make_a(const _Float16* p) {
  half8 lo = *(const half8*)p;
  half8 hi = *(const half8*)(p + 16);
  return __builtin_shufflevector(lo, hi, 0, 1, 2, 3, 4, 5, 6, 7,
                                 8, 9, 10, 11, 12, 13, 14, 15);
}

// LayerNorm one 16x512 tile; per-thread `vals` (32 elems, cols sub+16j of
// row `row`) already in registers. Writes f16 row into dstRow (stride-free).
__device__ __forceinline__ void ln_apply(const float* vals, int row, int sub,
                                         float* red, const float* lnw,
                                         const float* lnb, _Float16* dstRow) {
  float s = 0.f, s2 = 0.f;
  for (int j = 0; j < 32; ++j) {
    s += vals[j];
    s2 += vals[j] * vals[j];
  }
  __syncthreads();                       // red safe to overwrite
  red[row * 16 + sub] = s;
  red[256 + row * 16 + sub] = s2;
  __syncthreads();
  float ts = 0.f, ts2 = 0.f;
  for (int j = 0; j < 16; ++j) {
    ts += red[row * 16 + j];
    ts2 += red[256 + row * 16 + j];
  }
  float mu = ts * (1.0f / 512.0f);
  float var = ts2 * (1.0f / 512.0f) - mu * mu;
  float rstd = rsqrtf(var + 1e-5f);
  for (int j = 0; j < 32; ++j) {
    int c = sub + j * 16;
    dstRow[c] = (_Float16)((vals[j] - mu) * rstd * lnw[c] + lnb[c]);
  }
}

// ---------------------------------------------------------------------------
// Kernel 1: one workgroup (8 waves, 256 thr) per G=2 batch elements (M=32).
// ---------------------------------------------------------------------------
__global__ __launch_bounds__(256)
void fused_layer_kernel(const float* __restrict__ x,
                        const float* __restrict__ pe1,
                        const float* __restrict__ pe2,
                        const float* __restrict__ pos_emb,
                        const float* __restrict__ lnw_g,
                        const float* __restrict__ lnb_g,
                        const _Float16* __restrict__ wh,   // ws f16 weights
                        const float* __restrict__ ff1_b,
                        const float* __restrict__ prelu_a,
                        const float* __restrict__ ff2_w,   // [2][512]
                        const float* __restrict__ ff2_b,   // [2]
                        float* __restrict__ out) {
  __shared__ _Float16 abuf[G * 16 * 528];                 // t_in, later y
  // qh/kh while live; after softmax this region is reused as f32 attn-out.
  __shared__ __align__(64) unsigned char qkmem[2 * G * 16 * 528 * 2];
  __shared__ __align__(32) _Float16 vt[G * 512 * 16];     // V^T: [g][d][key]
  __shared__ _Float16 attnb[G * 8 * 16 * 40];             // probs (+K pad)
  __shared__ float lnw[512], lnb[512];
  __shared__ float red[2 * 16 * 16];
  __shared__ float out2[G * 32];
  __shared__ __align__(32) _Float16 zpad[32];             // zero B K-rows

  _Float16* qh = (_Float16*)qkmem;
  _Float16* kh = qh + G * 16 * 528;
  float* obuf = (float*)qkmem;                            // aliased later

  const int blk = blockIdx.x;                             // batches 2b, 2b+1
  const int tid = threadIdx.x;
  const int lane = tid & 31;
  const int wave = tid >> 5;
  const int jlane = lane & 15;
  const int glane = lane >> 4;

  if (tid < G * 32) out2[tid] = 0.f;
  if (tid < 32) zpad[tid] = (_Float16)0.f;
  for (int c = tid; c < 512; c += 256) {
    lnw[c] = lnw_g[c];
    lnb[c] = lnb_g[c];
  }

  // ---- Step A: t_in = LN(x + pos_emb + pe1 + pe2), fused in registers ----
  {
    int row = tid >> 4, sub = tid & 15;
    for (int g = 0; g < G; ++g) {
      const size_t base = ((size_t)(blk * G + g) * 16 + row) * DD;
      float vals[32];
      for (int j = 0; j < 32; ++j) {
        int c = sub + j * 16;
        vals[j] = x[base + c] + pe1[base + c] + pe2[base + c] +
                  pos_emb[row * DD + c];
      }
      ln_apply(vals, row, sub, red, lnw, lnb, abuf + (g * 16 + row) * 528);
    }
  }
  __syncthreads();

  // ---- Step B: QKV GEMMs. Each B tile feeds G WMMAs (M-tile reuse) ----
  for (int task = wave; task < 96; task += 8) {
    int mat = task / 32;
    int ntile = task % 32;
    int n = ntile * 16 + jlane;
    const _Float16* wb = wh + (size_t)mat * 262144 + (size_t)n * 512;
    v8f c0 = {}, c1 = {};
    for (int kb = 0; kb < 16; ++kb) {
      half16 bm = make_a(wb + kb * 32 + glane * 8);
      half16 a0 = make_a(abuf + jlane * 528 + kb * 32 + glane * 8);
      half16 a1 = make_a(abuf + (16 + jlane) * 528 + kb * 32 + glane * 8);
      c0 = __builtin_amdgcn_wmma_f32_16x16x32_f16(false, a0, false, bm,
                                                  (short)0, c0, false, false);
      c1 = __builtin_amdgcn_wmma_f32_16x16x32_f16(false, a1, false, bm,
                                                  (short)0, c1, false, false);
    }
    if (mat < 2) {
      _Float16* d = mat ? kh : qh;
      for (int r = 0; r < 8; ++r) {
        d[(r + 8 * glane) * 528 + n] = (_Float16)c0[r];
        d[(16 + r + 8 * glane) * 528 + n] = (_Float16)c1[r];
      }
    } else {                                   // V stored transposed [d][key]
      for (int r = 0; r < 8; ++r) {
        vt[(size_t)n * 16 + r + 8 * glane] = (_Float16)c0[r];
        vt[(size_t)(512 + n) * 16 + r + 8 * glane] = (_Float16)c1[r];
      }
    }
  }
  __syncthreads();

  // ---- Step C1: scores + softmax (head h = wave, both batches) ----
  {
    const int h = wave;
    for (int g = 0; g < G; ++g) {
      v8f sc = {};
      for (int kb = 0; kb < 2; ++kb) {
        half16 a = make_a(qh + (g * 16 + jlane) * 528 + h * 64 + kb * 32 + glane * 8);
        half16 bm = make_a(kh + (g * 16 + jlane) * 528 + h * 64 +
                           kb * 32 + glane * 8);
        sc = __builtin_amdgcn_wmma_f32_16x16x32_f16(false, a, false, bm,
                                                    (short)0, sc, false, false);
      }
      for (int r = 0; r < 8; ++r) {
        int i = r + 8 * glane;                 // query index
        float s = sc[r] * 0.125f;              // 1/sqrt(64)
        if (jlane > i) s = -1e9f;              // causal mask
        float mx = s;
        for (int m = 1; m < 16; m <<= 1) mx = fmaxf(mx, __shfl_xor(mx, m, 32));
        float e = __expf(s - mx);
        float sum = e;
        for (int m = 1; m < 16; m <<= 1) sum += __shfl_xor(sum, m, 32);
        float p = e / sum;
        attnb[((g * 8 + h) * 16 + i) * 40 + jlane] = (_Float16)p;
        attnb[((g * 8 + h) * 16 + i) * 40 + 16 + jlane] = (_Float16)0.f;
      }
    }
  }
  __syncthreads();   // qh/kh now dead -> region becomes f32 obuf

  // ---- Step C2: out_h = attn @ v_h  (M=16, K=16+16 zero pad, N=64) ----
  {
    const int h = wave;
    for (int g = 0; g < G; ++g) {
      half16 a = make_a(attnb + ((g * 8 + h) * 16 + jlane) * 40 + glane * 8);
      for (int nt = 0; nt < 4; ++nt) {
        int ncol = nt * 16 + jlane;            // d within head
        const _Float16* bp =
            vt + ((size_t)g * 512 + h * 64 + ncol) * 16 + glane * 8;
        half16 bm = __builtin_shufflevector(*(const half8*)bp, *(const half8*)zpad, 0, 1, 2, 3, 4, 5, 6, 7, 8, 9, 10, 11, 12, 13, 14, 15);        // elems 8..15: zero K rows
        v8f c = {};
        c = __builtin_amdgcn_wmma_f32_16x16x32_f16(false, a, false, bm,
                                                   (short)0, c, false, false);
        for (int r = 0; r < 8; ++r)
          obuf[(g * 16 + r + 8 * glane) * 516 + h * 64 + ncol] = c[r];
      }
    }
  }
  __syncthreads();

  // ---- Step D: y = LN(attn_out + x)  (x re-read from global/L2) ----
  {
    int row = tid >> 4, sub = tid & 15;
    for (int g = 0; g < G; ++g) {
      const size_t base = ((size_t)(blk * G + g) * 16 + row) * DD;
      float vals[32];
      for (int j = 0; j < 32; ++j) {
        int c = sub + j * 16;
        vals[j] = obuf[(g * 16 + row) * 516 + c] + x[base + c];
      }
      ln_apply(vals, row, sub, red, lnw, lnb, abuf + (g * 16 + row) * 528);
    }
  }
  __syncthreads();

  // ---- Step E: FF1 + PReLU fused with FF2 reduction ----
  {
    const _Float16* ff1h = wh + (size_t)3 * 262144;
    const float pa = *prelu_a;
    float acc[G][2][8];
    for (int g = 0; g < G; ++g)
      for (int j = 0; j < 2; ++j)
        for (int r = 0; r < 8; ++r) acc[g][j][r] = 0.f;

    for (int nt = wave * 4; nt < wave * 4 + 4; ++nt) {
      int n = nt * 16 + jlane;
      const _Float16* wb = ff1h + (size_t)n * 512;
      v8f c0 = {}, c1 = {};
      for (int kb = 0; kb < 16; ++kb) {
        half16 bm = make_a(wb + kb * 32 + glane * 8);
        half16 a0 = make_a(abuf + jlane * 528 + kb * 32 + glane * 8);
        half16 a1 = make_a(abuf + (16 + jlane) * 528 + kb * 32 + glane * 8);
        c0 = __builtin_amdgcn_wmma_f32_16x16x32_f16(false, a0, false, bm,
                                                    (short)0, c0, false, false);
        c1 = __builtin_amdgcn_wmma_f32_16x16x32_f16(false, a1, false, bm,
                                                    (short)0, c1, false, false);
      }
      float b1 = ff1_b[n];
      float w20 = ff2_w[n];
      float w21 = ff2_w[512 + n];
      for (int r = 0; r < 8; ++r) {
        float h0 = c0[r] + b1, h1 = c1[r] + b1;
        h0 = h0 > 0.f ? h0 : pa * h0;          // PReLU
        h1 = h1 > 0.f ? h1 : pa * h1;
        acc[0][0][r] += h0 * w20;
        acc[0][1][r] += h0 * w21;
        acc[1][0][r] += h1 * w20;
        acc[1][1][r] += h1 * w21;
      }
    }
    for (int m = 1; m < 16; m <<= 1)
      for (int g = 0; g < G; ++g)
        for (int j = 0; j < 2; ++j)
          for (int r = 0; r < 8; ++r)
            acc[g][j][r] += __shfl_xor(acc[g][j][r], m, 32);
    if (jlane == 0) {
      for (int g = 0; g < G; ++g)
        for (int r = 0; r < 8; ++r) {
          int m = r + 8 * glane;
          atomicAdd(&out2[g * 32 + m * 2 + 0], acc[g][0][r]);
          atomicAdd(&out2[g * 32 + m * 2 + 1], acc[g][1][r]);
        }
    }
  }
  __syncthreads();

  if (tid < G * 32) {
    int g = tid >> 5, rem = tid & 31;
    int m = rem >> 1, j = rem & 1;
    out[(((size_t)(blk * G + g)) * 16 + m) * 2 + j] = out2[tid] + ff2_b[j];
  }
}

extern "C" void kernel_launch(void* const* d_in, const int* in_sizes, int n_in,
                              void* d_out, int out_size, void* d_ws, size_t ws_size,
                              hipStream_t stream) {
  (void)in_sizes; (void)n_in; (void)out_size; (void)ws_size;
  const float* x    = (const float*)d_in[0];
  const float* pe1  = (const float*)d_in[1];
  const float* pe2  = (const float*)d_in[2];
  /* d_in[3] exist_nodes: unused by the reference */
  const float* pos  = (const float*)d_in[4];
  const float* Qw   = (const float*)d_in[5];
  const float* Kw   = (const float*)d_in[6];
  const float* Vw   = (const float*)d_in[7];
  const float* lnw  = (const float*)d_in[8];
  const float* lnb  = (const float*)d_in[9];
  const float* ff1w = (const float*)d_in[10];
  const float* ff1b = (const float*)d_in[11];
  const float* pa   = (const float*)d_in[12];
  const float* ff2w = (const float*)d_in[13];
  const float* ff2b = (const float*)d_in[14];
  _Float16* wsh = (_Float16*)d_ws;               // 4 * 512KB f16 weights

  convert_weights_kernel<<<4096, 256, 0, stream>>>(Qw, Kw, Vw, ff1w, wsh);
  fused_layer_kernel<<<NWG, 256, 0, stream>>>(x, pe1, pe2, pos, lnw, lnb, wsh,
                                              ff1b, pa, ff2w, ff2b,
                                              (float*)d_out);
}
